// TransformerBlock_37735582663062
// MI455X (gfx1250) — matrix operand order repaired and checked by reading, not compiled, not run on hardware
//
#include <hip/hip_runtime.h>

// ---------------------------------------------------------------------------
// Transformer block for MI455X (gfx1250): bf16 WMMA, TDM-staged A tiles,
// register-pipelined B fragments, fp32 softmax/norms.
// ---------------------------------------------------------------------------

#define DIMC    2048
#define NHEAD   16
#define HDIM    128
#define HIDDENC 8192
#define SEQ     2048
#define BATCH   2
#define TOKENS  (BATCH * SEQ)   // 4096

typedef __attribute__((ext_vector_type(16))) __bf16   v16bf;
typedef __attribute__((ext_vector_type(8)))  __bf16   v8bf;
typedef __attribute__((ext_vector_type(4)))  __bf16   v4bf;
typedef __attribute__((ext_vector_type(8)))  float    v8f;
typedef __attribute__((ext_vector_type(4)))  float    v4f;
typedef __attribute__((ext_vector_type(4)))  unsigned u32x4;
typedef __attribute__((ext_vector_type(4)))  int      i32x4;
typedef __attribute__((ext_vector_type(8)))  int      i32x8;

#if defined(__AMDGCN__) && __has_builtin(__builtin_amdgcn_tensor_load_to_lds) && \
    __has_builtin(__builtin_amdgcn_s_wait_tensorcnt)
#define HAVE_TDM 1
#else
#define HAVE_TDM 0
#endif

// ---- fragment loaders (NT layout per CDNA5 ISA 7.12.2) --------------------
// A (16x32 bf16): lane<16 -> K {0..7, 16..23}; lane>=16 -> K {8..15, 24..31}
static __device__ inline v16bf load_frag_a(const __bf16* base, int ld,
                                           int row0, int k0, int lane) {
  int r  = row0 + (lane & 15);
  int kk = k0 + ((lane >> 4) << 3);
  const __bf16* p = base + (size_t)r * ld + kk;
  v8bf lo = *(const v8bf*)p;
  v8bf hi = *(const v8bf*)(p + 16);
  return __builtin_shufflevector(lo, hi, 0,1,2,3,4,5,6,7,8,9,10,11,12,13,14,15);
}

// A fragment out of an LDS-resident 128x32 tile (row0 relative to tile)
static __device__ inline v16bf lds_frag_a(const __bf16* tile, int row0, int lane) {
  int r  = row0 + (lane & 15);
  int kk = (lane >> 4) << 3;
  const __bf16* p = tile + r * 32 + kk;
  v8bf lo = *(const v8bf*)p;
  v8bf hi = *(const v8bf*)(p + 16);
  return __builtin_shufflevector(lo, hi, 0,1,2,3,4,5,6,7,8,9,10,11,12,13,14,15);
}

// B (32x16 bf16): column n = lane&15; lane<16 -> K {0..7, 16..23}; lane>=16 -> K {8..15, 24..31}
static __device__ inline v16bf load_frag_b(const __bf16* base, int ld,
                                           int n0, int k0, int lane) {
  int n  = n0 + (lane & 15);
  int kk = k0 + ((lane >> 4) << 3);
  const __bf16* p = base + (size_t)n * ld + kk;
  v8bf lo = *(const v8bf*)p;
  v8bf hi = *(const v8bf*)(p + 16);
  return __builtin_shufflevector(lo, hi, 0,1,2,3,4,5,6,7,8,9,10,11,12,13,14,15);
}

static __device__ inline v8f wmma_bf16(v16bf a, v16bf b, v8f c) {
  return __builtin_amdgcn_wmma_f32_16x16x32_bf16(false, a, false, b,
                                                 (short)0, c, false, false);
}

// ---- TDM: 2-D tile load (rows x 32 bf16 cols) global -> LDS ---------------
// D# per CDNA5 ISA ch.8: group0 {count|lds_addr|global_addr|type=2},
// group1 {data_size=2B, tensor dims, tile dims, dim0 stride}.
static __device__ inline void tdm_load_tile_2d(unsigned lds_addr,
                                               const __bf16* gptr,
                                               unsigned tensor_d0, unsigned tensor_d1,
                                               unsigned tile_d0, unsigned tile_d1,
                                               unsigned stride0) {
#if HAVE_TDM
  unsigned long long ga = (unsigned long long)(uintptr_t)gptr;
  u32x4 g0 = { 1u,                                   // count=1 (user descriptor)
               lds_addr,                              // [63:32] LDS byte address
               (unsigned)(ga & 0xFFFFFFFFu),          // [95:64] global addr lo
               (unsigned)((ga >> 32) & 0x01FFFFFFu) | (2u << 30) }; // type=2
  i32x8 g1 = { (int)(1u << 16),                                        // data_size=1 -> 2B
               (int)((tensor_d0 & 0xFFFFu) << 16),                     // dim0[15:0]
               (int)(((tensor_d0 >> 16) & 0xFFFFu) |
                     ((tensor_d1 & 0xFFFFu) << 16)),                   // dim0 hi | dim1 lo
               (int)(((tensor_d1 >> 16) & 0xFFFFu) |
                     ((tile_d0 & 0xFFFFu) << 16)),                     // dim1 hi | tile0
               (int)(tile_d1 & 0xFFFFu),                               // tile1 (tile2=0)
               (int)stride0,                                           // dim0 stride lo
               0, 0 };
  i32x4 z4 = { 0, 0, 0, 0 };
#if __clang_major__ >= 23
  i32x8 z8 = { 0, 0, 0, 0, 0, 0, 0, 0 };
  __builtin_amdgcn_tensor_load_to_lds(g0, g1, z4, z4, z8, 0);
#else
  __builtin_amdgcn_tensor_load_to_lds(g0, g1, z4, z4, 0);
#endif
#else
  (void)lds_addr; (void)gptr; (void)tensor_d0; (void)tensor_d1;
  (void)tile_d0; (void)tile_d1; (void)stride0;
#endif
}

// ---- fp32 -> bf16 conversion (x4 vectorized) ------------------------------
__global__ void f32_to_bf16_k(const float* __restrict__ s,
                              __bf16* __restrict__ d, size_t n4) {
  size_t i = (size_t)blockIdx.x * 256 + threadIdx.x;
  if (i >= n4) return;
  v4f v = *(const v4f*)(s + i * 4);
  v4bf o = { (__bf16)v.x, (__bf16)v.y, (__bf16)v.z, (__bf16)v.w };
  *(v4bf*)(d + i * 4) = o;
}

// ---- RMSNorm (fp32 in, bf16 out) ------------------------------------------
__global__ __launch_bounds__(256)
void rmsnorm_bf16_k(const float* __restrict__ x, const float* __restrict__ scale,
                    __bf16* __restrict__ out) {
  int t = blockIdx.x;
  const float* xp = x + (size_t)t * DIMC;
  float ss = 0.f;
  for (int i = threadIdx.x; i < DIMC; i += 256) { float v = xp[i]; ss += v * v; }
#pragma unroll
  for (int off = 16; off > 0; off >>= 1) ss += __shfl_xor(ss, off, 32);
  __shared__ float red[8];
  if ((threadIdx.x & 31) == 0) red[threadIdx.x >> 5] = ss;
  __syncthreads();
  float tot = 0.f;
#pragma unroll
  for (int w = 0; w < 8; ++w) tot += red[w];
  float inv = rsqrtf(tot / (float)DIMC + 1e-6f);
  for (int i = threadIdx.x; i < DIMC; i += 256)
    out[(size_t)t * DIMC + i] = (__bf16)(scale[i] * xp[i] * inv);
}

// ---- NT GEMM: C[M,N] = A[M,K] * B[N,K]^T ----------------------------------
// Block tile 128M x 128N, 8 waves (4 M-waves x 32 rows, 2 N-waves x 64 cols).
// A tile staged to LDS by the Tensor Data Mover (double-buffered); B fragments
// register-pipelined. EPI 0: bf16 store. EPI 1: f32 store = acc + RES.
template <int EPI>
__global__ __launch_bounds__(256)
void gemm_nt_k(const __bf16* __restrict__ A, const __bf16* __restrict__ Bw,
               const float* __restrict__ RES, void* __restrict__ Cout,
               int M, int N, int K) {
#if HAVE_TDM
  __shared__ __bf16 aTile[2][128 * 32];   // 2 x 8KB
#endif
  int lane = threadIdx.x & 31;
  int wid  = threadIdx.x >> 5;
  int wm   = wid & 3;                      // 4 M-waves
  int wn   = wid >> 2;                     // 2 N-waves
  int mblk = blockIdx.y * 128;
  int n0   = blockIdx.x * 128 + wn * 64;

  v8f acc[2][4] = {};
  v16bf bcur[4], bnext[4];

#if HAVE_TDM
  unsigned aBase = (unsigned)(uintptr_t)&aTile[0][0];
  const unsigned TBYTES = 128 * 32 * 2;
  if (wid == 0) {
    tdm_load_tile_2d(aBase, A + (size_t)mblk * K, (unsigned)K, (unsigned)M,
                     32u, 128u, (unsigned)K);
    __builtin_amdgcn_s_wait_tensorcnt(0);
  }
  __syncthreads();
#endif
#pragma unroll
  for (int t = 0; t < 4; ++t) bcur[t] = load_frag_b(Bw, K, n0 + t * 16, 0, lane);

  int buf = 0;
  for (int k0 = 0; k0 < K; k0 += 32) {
    int  kn   = k0 + 32;
    bool last = (kn >= K);
#if HAVE_TDM
    if (!last && wid == 0)
      tdm_load_tile_2d(aBase + (unsigned)(buf ^ 1) * TBYTES,
                       A + (size_t)mblk * K + kn, (unsigned)K, (unsigned)M,
                       32u, 128u, (unsigned)K);
#endif
    // pipeline next B fragments while this step's WMMAs run
    int kb = last ? k0 : kn;
#pragma unroll
    for (int t = 0; t < 4; ++t) bnext[t] = load_frag_b(Bw, K, n0 + t * 16, kb, lane);
    __builtin_prefetch((const void*)(Bw + (size_t)(n0 + (lane & 15)) * K + k0 + 128), 0, 1);

#if HAVE_TDM
    v16bf a0 = lds_frag_a(&aTile[buf][0], wm * 32,      lane);
    v16bf a1 = lds_frag_a(&aTile[buf][0], wm * 32 + 16, lane);
#else
    v16bf a0 = load_frag_a(A, K, mblk + wm * 32,      k0, lane);
    v16bf a1 = load_frag_a(A, K, mblk + wm * 32 + 16, k0, lane);
#endif
#pragma unroll
    for (int t = 0; t < 4; ++t) {
      acc[0][t] = wmma_bf16(a0, bcur[t], acc[0][t]);
      acc[1][t] = wmma_bf16(a1, bcur[t], acc[1][t]);
    }
#pragma unroll
    for (int t = 0; t < 4; ++t) bcur[t] = bnext[t];
#if HAVE_TDM
    if (!last) {
      if (wid == 0) __builtin_amdgcn_s_wait_tensorcnt(0);
      __syncthreads();
      buf ^= 1;
    }
#endif
  }

  int roff = (lane >> 4) << 3;
#pragma unroll
  for (int sub = 0; sub < 2; ++sub)
#pragma unroll
    for (int t = 0; t < 4; ++t)
#pragma unroll
      for (int i = 0; i < 8; ++i) {
        int row = mblk + wm * 32 + sub * 16 + i + roff;
        int col = n0 + t * 16 + (lane & 15);
        size_t idx = (size_t)row * N + col;
        if (EPI == 0) ((__bf16*)Cout)[idx] = (__bf16)acc[sub][t][i];
        else          ((float*)Cout)[idx]  = acc[sub][t][i] + RES[idx];
      }
}

// ---- fused dual GEMM + SwiGLU: out = silu(A*B1^T) * (A*B2^T), bf16 --------
// Same 128M x 128N tiling as gemm_nt_k, TDM-staged A tile, dual accumulators.
__global__ __launch_bounds__(256)
void gemm_nt_swiglu_k(const __bf16* __restrict__ A, const __bf16* __restrict__ B1,
                      const __bf16* __restrict__ B2, __bf16* __restrict__ Cout,
                      int M, int N, int K) {
#if HAVE_TDM
  __shared__ __bf16 aTile[2][128 * 32];   // 2 x 8KB
#endif
  int lane = threadIdx.x & 31;
  int wid  = threadIdx.x >> 5;
  int wm   = wid & 3;
  int wn   = wid >> 2;
  int mblk = blockIdx.y * 128;
  int n0   = blockIdx.x * 128 + wn * 64;

  v8f accG[2][4] = {}, accV[2][4] = {};

#if HAVE_TDM
  unsigned aBase = (unsigned)(uintptr_t)&aTile[0][0];
  const unsigned TBYTES = 128 * 32 * 2;
  if (wid == 0) {
    tdm_load_tile_2d(aBase, A + (size_t)mblk * K, (unsigned)K, (unsigned)M,
                     32u, 128u, (unsigned)K);
    __builtin_amdgcn_s_wait_tensorcnt(0);
  }
  __syncthreads();
#endif

  int buf = 0;
  for (int k0 = 0; k0 < K; k0 += 32) {
    int  kn   = k0 + 32;
    bool last = (kn >= K);
#if HAVE_TDM
    if (!last && wid == 0)
      tdm_load_tile_2d(aBase + (unsigned)(buf ^ 1) * TBYTES,
                       A + (size_t)mblk * K + kn, (unsigned)K, (unsigned)M,
                       32u, 128u, (unsigned)K);
    v16bf a0 = lds_frag_a(&aTile[buf][0], wm * 32,      lane);
    v16bf a1 = lds_frag_a(&aTile[buf][0], wm * 32 + 16, lane);
#else
    v16bf a0 = load_frag_a(A, K, mblk + wm * 32,      k0, lane);
    v16bf a1 = load_frag_a(A, K, mblk + wm * 32 + 16, k0, lane);
#endif
    __builtin_prefetch((const void*)(B1 + (size_t)(n0 + (lane & 15)) * K + k0 + 128), 0, 1);
    __builtin_prefetch((const void*)(B2 + (size_t)(n0 + (lane & 15)) * K + k0 + 128), 0, 1);
#pragma unroll
    for (int t = 0; t < 4; ++t) {
      v16bf b1 = load_frag_b(B1, K, n0 + t * 16, k0, lane);
      accG[0][t] = wmma_bf16(a0, b1, accG[0][t]);
      accG[1][t] = wmma_bf16(a1, b1, accG[1][t]);
      v16bf b2 = load_frag_b(B2, K, n0 + t * 16, k0, lane);
      accV[0][t] = wmma_bf16(a0, b2, accV[0][t]);
      accV[1][t] = wmma_bf16(a1, b2, accV[1][t]);
    }
#if HAVE_TDM
    if (!last) {
      if (wid == 0) __builtin_amdgcn_s_wait_tensorcnt(0);
      __syncthreads();
      buf ^= 1;
    }
#endif
  }

  int roff = (lane >> 4) << 3;
#pragma unroll
  for (int sub = 0; sub < 2; ++sub)
#pragma unroll
    for (int t = 0; t < 4; ++t)
#pragma unroll
      for (int i = 0; i < 8; ++i) {
        int row = mblk + wm * 32 + sub * 16 + i + roff;
        int col = n0 + t * 16 + (lane & 15);
        float g = accG[sub][t][i];
        float s = g / (1.f + __expf(-g));    // silu
        Cout[(size_t)row * N + col] = (__bf16)(s * accV[sub][t][i]);
      }
}

// ---- V transpose: Vt[b][h][d][t] = V[(b*T+t)*DIM + h*128 + d] -------------
__global__ void transpose_v_k(const __bf16* __restrict__ V,
                              __bf16* __restrict__ Vt) {
  size_t i = (size_t)blockIdx.x * 256 + threadIdx.x;
  const size_t total = (size_t)BATCH * NHEAD * HDIM * SEQ;
  if (i >= total) return;
  int t = (int)(i % SEQ); size_t r = i / SEQ;
  int d = (int)(r % HDIM); r /= HDIM;
  int h = (int)(r % NHEAD); int b = (int)(r / NHEAD);
  Vt[i] = V[((size_t)(b * SEQ + t)) * DIMC + h * HDIM + d];
}

// ---- causal flash attention: 1 wave per (b, h, 16 query rows) -------------
__global__ __launch_bounds__(32)
void attn_fwd_k(const __bf16* __restrict__ Q, const __bf16* __restrict__ K,
                const __bf16* __restrict__ Vt, __bf16* __restrict__ O) {
  __shared__ __bf16 pLds[16 * 32];
  int lane = threadIdx.x & 31;
  int q0 = blockIdx.x * 16;
  int h  = blockIdx.y;
  int b  = blockIdx.z;
  const __bf16* qp = Q + ((size_t)b * SEQ) * DIMC + h * HDIM;
  const __bf16* kp = K + ((size_t)b * SEQ) * DIMC + h * HDIM;
  const __bf16* vp = Vt + ((size_t)(b * NHEAD + h)) * HDIM * SEQ;

  v16bf aq[4];
#pragma unroll
  for (int c = 0; c < 4; ++c) aq[c] = load_frag_a(qp, DIMC, q0, c * 32, lane);

  v8f o[8] = {};
  float mrow[8], lrow[8];
#pragma unroll
  for (int i = 0; i < 8; ++i) { mrow[i] = -1e30f; lrow[i] = 0.f; }
  const float sc = 0.08838834764831845f;       // 1/sqrt(128)
  int roff = (lane >> 4) << 3;

  for (int k0 = 0; k0 < q0 + 16; k0 += 32) {   // 32-key tiles, causal bound
    v8f s0 = {}, s1 = {};
#pragma unroll
    for (int c = 0; c < 4; ++c) {              // head dim 128 = 4 x 32
      v16bf bk0 = load_frag_b(kp, DIMC, k0,      c * 32, lane);
      v16bf bk1 = load_frag_b(kp, DIMC, k0 + 16, c * 32, lane);
      s0 = wmma_bf16(aq[c], bk0, s0);
      s1 = wmma_bf16(aq[c], bk1, s1);
    }
    // online softmax; row m = i + roff lives in VGPR i across a 16-lane half
#pragma unroll
    for (int i = 0; i < 8; ++i) {
      int row = q0 + i + roff;
      int c0  = k0 + (lane & 15);
      float x0 = (c0      <= row) ? s0[i] * sc : -1e30f;
      float x1 = (c0 + 16 <= row) ? s1[i] * sc : -1e30f;
      float m = fmaxf(x0, x1);
#pragma unroll
      for (int off = 1; off < 16; off <<= 1) m = fmaxf(m, __shfl_xor(m, off, 32));
      float mn   = fmaxf(mrow[i], m);
      float corr = __expf(mrow[i] - mn);
      mrow[i] = mn;
      float p0 = __expf(x0 - mn);
      float p1 = __expf(x1 - mn);
      float rs = p0 + p1;
#pragma unroll
      for (int off = 1; off < 16; off <<= 1) rs += __shfl_xor(rs, off, 32);
      lrow[i] = lrow[i] * corr + rs;
#pragma unroll
      for (int j = 0; j < 8; ++j) o[j][i] *= corr;
      int ri = i + roff;
      pLds[ri * 32 + (lane & 15)]      = (__bf16)p0;
      pLds[ri * 32 + 16 + (lane & 15)] = (__bf16)p1;
    }
    __syncthreads();
    // re-pack P from C-layout (LDS) into A-fragment layout
    int r  = lane & 15;
    int kk = (lane >> 4) << 3;
    v8bf plo = *(const v8bf*)&pLds[r * 32 + kk];
    v8bf phi = *(const v8bf*)&pLds[r * 32 + 16 + kk];
    v16bf ap = __builtin_shufflevector(plo, phi,
                 0,1,2,3,4,5,6,7,8,9,10,11,12,13,14,15);
    __syncthreads();
#pragma unroll
    for (int j = 0; j < 8; ++j) {              // O 16x128 = 8 N-tiles
      v16bf bv = load_frag_b(vp, SEQ, j * 16, k0, lane);
      o[j] = wmma_bf16(ap, bv, o[j]);
    }
  }
#pragma unroll
  for (int i = 0; i < 8; ++i) {
    float inv = 1.f / lrow[i];
    int row = q0 + i + roff;
#pragma unroll
    for (int j = 0; j < 8; ++j) {
      int col = h * HDIM + j * 16 + (lane & 15);
      O[((size_t)(b * SEQ + row)) * DIMC + col] = (__bf16)(o[j][i] * inv);
    }
  }
}

// ---------------------------------------------------------------------------
extern "C" void kernel_launch(void* const* d_in, const int* in_sizes, int n_in,
                              void* d_out, int out_size, void* d_ws, size_t ws_size,
                              hipStream_t stream) {
  const float* x   = (const float*)d_in[0];
  const float* wq  = (const float*)d_in[1];
  const float* wk  = (const float*)d_in[2];
  const float* wv  = (const float*)d_in[3];
  const float* wo  = (const float*)d_in[4];
  const float* w1  = (const float*)d_in[5];
  const float* w2  = (const float*)d_in[6];
  const float* w3  = (const float*)d_in[7];
  const float* sc1 = (const float*)d_in[8];
  const float* sc2 = (const float*)d_in[9];

  char* ws = (char*)d_ws;
  size_t off = 0;
  auto alloc = [&](size_t bytes) -> void* {
    void* p = ws + off;
    off += (bytes + 255) & ~(size_t)255;
    return p;
  };
  const size_t WD  = (size_t)DIMC * DIMC;
  const size_t WH  = (size_t)HIDDENC * DIMC;
  const size_t ACT = (size_t)TOKENS * DIMC;

  __bf16* wq_b = (__bf16*)alloc(WD * 2);
  __bf16* wk_b = (__bf16*)alloc(WD * 2);
  __bf16* wv_b = (__bf16*)alloc(WD * 2);
  __bf16* wo_b = (__bf16*)alloc(WD * 2);
  __bf16* w1_b = (__bf16*)alloc(WH * 2);
  __bf16* w2_b = (__bf16*)alloc(WH * 2);
  __bf16* w3_b = (__bf16*)alloc(WH * 2);
  __bf16* xn1  = (__bf16*)alloc(ACT * 2);
  __bf16* Qb   = (__bf16*)alloc(ACT * 2);
  __bf16* Kb   = (__bf16*)alloc(ACT * 2);
  __bf16* Vb   = (__bf16*)alloc(ACT * 2);
  __bf16* Vtb  = (__bf16*)alloc(ACT * 2);
  __bf16* Ob   = (__bf16*)alloc(ACT * 2);
  float*  x1   = (float*) alloc(ACT * 4);
  __bf16* xn2  = (__bf16*)alloc(ACT * 2);
  __bf16* hmid = (__bf16*)alloc((size_t)TOKENS * HIDDENC * 2);

  // weights fp32 -> bf16 (vectorized x4)
  f32_to_bf16_k<<<(int)((WD / 4 + 255) / 256), 256, 0, stream>>>(wq, wq_b, WD / 4);
  f32_to_bf16_k<<<(int)((WD / 4 + 255) / 256), 256, 0, stream>>>(wk, wk_b, WD / 4);
  f32_to_bf16_k<<<(int)((WD / 4 + 255) / 256), 256, 0, stream>>>(wv, wv_b, WD / 4);
  f32_to_bf16_k<<<(int)((WD / 4 + 255) / 256), 256, 0, stream>>>(wo, wo_b, WD / 4);
  f32_to_bf16_k<<<(int)((WH / 4 + 255) / 256), 256, 0, stream>>>(w1, w1_b, WH / 4);
  f32_to_bf16_k<<<(int)((WH / 4 + 255) / 256), 256, 0, stream>>>(w2, w2_b, WH / 4);
  f32_to_bf16_k<<<(int)((WH / 4 + 255) / 256), 256, 0, stream>>>(w3, w3_b, WH / 4);

  // x -> rmsnorm -> bf16
  rmsnorm_bf16_k<<<TOKENS, 256, 0, stream>>>(x, sc1, xn1);

  // Q/K/V projections (block tile 128x128)
  dim3 gD(DIMC / 128, TOKENS / 128);
  gemm_nt_k<0><<<gD, 256, 0, stream>>>(xn1, wq_b, nullptr, Qb, TOKENS, DIMC, DIMC);
  gemm_nt_k<0><<<gD, 256, 0, stream>>>(xn1, wk_b, nullptr, Kb, TOKENS, DIMC, DIMC);
  gemm_nt_k<0><<<gD, 256, 0, stream>>>(xn1, wv_b, nullptr, Vb, TOKENS, DIMC, DIMC);

  // V -> Vt[b][h][d][t]
  const size_t VT = (size_t)BATCH * NHEAD * HDIM * SEQ;
  transpose_v_k<<<(int)((VT + 255) / 256), 256, 0, stream>>>(Vb, Vtb);

  // flash attention
  attn_fwd_k<<<dim3(SEQ / 16, NHEAD, BATCH), 32, 0, stream>>>(Qb, Kb, Vtb, Ob);

  // output projection + residual (fp32)
  gemm_nt_k<1><<<gD, 256, 0, stream>>>(Ob, wo_b, x, x1, TOKENS, DIMC, DIMC);

  // second norm
  rmsnorm_bf16_k<<<TOKENS, 256, 0, stream>>>(x1, sc2, xn2);

  // fused SwiGLU dual GEMM (block tile 128x128, TDM A staging)
  dim3 gH(HIDDENC / 128, TOKENS / 128);
  gemm_nt_swiglu_k<<<gH, 256, 0, stream>>>(xn2, w1_b, w2_b, hmid,
                                           TOKENS, HIDDENC, DIMC);

  // down projection + residual -> fp32 output
  gemm_nt_k<1><<<gD, 256, 0, stream>>>(hmid, w3_b, x1, (float*)d_out,
                                       TOKENS, DIMC, HIDDENC);
}
